// TransformerBlock_84207128805915
// MI455X (gfx1250) — compile-verified, matrix operand order checked by reading, not run on hardware
//
#include <hip/hip_runtime.h>
#include <math.h>

typedef __attribute__((ext_vector_type(16))) __bf16 v16bf;
typedef __attribute__((ext_vector_type(8)))  float  v8f;
typedef __attribute__((ext_vector_type(4)))  unsigned int u32x4;
typedef __attribute__((ext_vector_type(8)))  unsigned int u32x8;

#define DIM   768
#define NH    12
#define HD    64
#define WS14  14
#define LQ    196
#define LP    224           // 196 padded to multiple of 32
#define NWIN  200           // 8 images * 5*5 windows
#define MQ    (NWIN * LP)   // 44800 window-token rows (multiple of 128)
#define MT    32768         // 8*64*64 pixel tokens
#define MLPH  3072

static __device__ __forceinline__ unsigned short f2bf(float f) {
  return __builtin_bit_cast(unsigned short, (__bf16)f);   // native RNE cvt
}
static __device__ __forceinline__ unsigned int pack2bf(float lo, float hi) {
  typedef __attribute__((ext_vector_type(2))) __bf16 v2bf_t;
  v2bf_t p;
  p[0] = (__bf16)lo;
  p[1] = (__bf16)hi;
  return __builtin_bit_cast(unsigned int, p);             // -> v_cvt_pk_bf16_f32
}
static __device__ __forceinline__ float bf2f(unsigned short u) {
  unsigned int x = ((unsigned int)u) << 16;
  return __builtin_bit_cast(float, x);
}
static __device__ __forceinline__ __bf16 us2bf(unsigned short u) {
  return __builtin_bit_cast(__bf16, u);
}
static __device__ __forceinline__ v8f zero_v8f() {
  v8f z;
#pragma unroll
  for (int i = 0; i < 8; ++i) z[i] = 0.0f;
  return z;
}
// ISA 7.12.2: 16-bit A/B fragment K-pair base for VGPR j, given lane
static __device__ __forceinline__ int frag_k(int j, int lane) {
  return 2 * (j & 3) + ((lane >= 16) ? 8 : 0) + ((j >= 4) ? 16 : 0);
}

// ---------------------------------------------------------------------------
// Tensor Data Mover: async 2D tile load (bf16/u16 elements) global -> LDS.
// D# per CDNA5 ISA: group0 = {count/lds/global},
// group1 = {data_size, tensor dims = tile dims (tile always in-bounds),
// tile dims, dim0 stride}. 2-group form (<=2D), issued wave-uniform.
// ---------------------------------------------------------------------------
static __device__ __forceinline__ void tdm_load_tile_b16(
    const void* gptr, unsigned lds_byte_addr, unsigned tile_x_elems,
    unsigned tile_y_rows, unsigned row_stride_elems) {
  unsigned long long ga = (unsigned long long)gptr;
  u32x4 g0;
  g0[0] = 1u;                                   // count=1, user descriptor
  g0[1] = lds_byte_addr;                        // LDS address (bytes)
  g0[2] = (unsigned int)ga;                     // global addr [31:0]
  g0[3] = (unsigned int)(ga >> 32) | (2u << 30);// addr[56:32] | type=2 (image)
  u32x8 g1;
  g1[0] = 0x00010000u;                          // data_size=1 (2 bytes)
  g1[1] = (tile_x_elems & 0xFFFFu) << 16;       // tensor_dim0[15:0] @bits63:48
  g1[2] = (tile_x_elems >> 16) | (tile_y_rows << 16); // dim0 hi | dim1 lo
  g1[3] = (tile_y_rows >> 16) | (tile_x_elems << 16); // dim1 hi | tile_dim0
  g1[4] = tile_y_rows;                          // tile_dim1 (tile_dim2=0)
  g1[5] = row_stride_elems;                     // tensor_dim0_stride [31:0]
  g1[6] = 0u;
  g1[7] = 0u;
  asm volatile("tensor_load_to_lds %0, %1" ::"s"(g0), "s"(g1) : "memory");
}

// ---------------------------------------------------------------------------
// LN1 fused with window partition: x (8,64,64,768) f32 -> hwin (MQ,768) bf16
// ---------------------------------------------------------------------------
__global__ __launch_bounds__(256) void ln1_window_kernel(
    const float* __restrict__ x, const float* __restrict__ w,
    const float* __restrict__ b, unsigned short* __restrict__ hwin) {
  int token = blockIdx.x;            // 0..MQ-1
  int win = token / LP, t = token % LP;
  int bimg = win / 25, wh = (win % 25) / 5, ww = win % 5;
  int hh = wh * WS14 + t / WS14;
  int cc = ww * WS14 + t % WS14;
  bool valid = (t < LQ) && (hh < 64) && (cc < 64);
  const float* row = x + (((size_t)bimg * 64 + hh) * 64 + cc) * DIM;

  __shared__ float red[256];
  int tid = threadIdx.x;
  float s = 0.f;
  if (valid)
    for (int i = tid; i < DIM; i += 256) s += row[i];
  red[tid] = s;
  for (int st = 128; st > 0; st >>= 1) {
    __syncthreads();
    if (tid < st) red[tid] += red[tid + st];
  }
  __syncthreads();
  float mu = red[0] * (1.f / DIM);
  __syncthreads();
  float v = 0.f;
  if (valid)
    for (int i = tid; i < DIM; i += 256) { float d = row[i] - mu; v += d * d; }
  red[tid] = v;
  for (int st = 128; st > 0; st >>= 1) {
    __syncthreads();
    if (tid < st) red[tid] += red[tid + st];
  }
  __syncthreads();
  float rinv = rsqrtf(red[0] * (1.f / DIM) + 1e-5f);
  unsigned short* orow = hwin + (size_t)token * DIM;
  for (int i = tid; i < DIM; i += 256) {
    float val = valid ? ((row[i] - mu) * rinv * w[i] + b[i]) : 0.f;
    orow[i] = f2bf(val);
  }
}

// ---------------------------------------------------------------------------
// LN2 (plain): xres (MT,768) f32 -> h2 (MT,768) bf16
// ---------------------------------------------------------------------------
__global__ __launch_bounds__(256) void ln2_kernel(
    const float* __restrict__ xin, const float* __restrict__ w,
    const float* __restrict__ b, unsigned short* __restrict__ out) {
  int token = blockIdx.x;
  const float* row = xin + (size_t)token * DIM;
  __shared__ float red[256];
  int tid = threadIdx.x;
  float s = 0.f;
  for (int i = tid; i < DIM; i += 256) s += row[i];
  red[tid] = s;
  for (int st = 128; st > 0; st >>= 1) {
    __syncthreads();
    if (tid < st) red[tid] += red[tid + st];
  }
  __syncthreads();
  float mu = red[0] * (1.f / DIM);
  __syncthreads();
  float v = 0.f;
  for (int i = tid; i < DIM; i += 256) { float d = row[i] - mu; v += d * d; }
  red[tid] = v;
  for (int st = 128; st > 0; st >>= 1) {
    __syncthreads();
    if (tid < st) red[tid] += red[tid + st];
  }
  __syncthreads();
  float rinv = rsqrtf(red[0] * (1.f / DIM) + 1e-5f);
  unsigned short* orow = out + (size_t)token * DIM;
  for (int i = tid; i < DIM; i += 256)
    orow[i] = f2bf((row[i] - mu) * rinv * w[i] + b[i]);
}

// ---------------------------------------------------------------------------
// Generic bf16 WMMA GEMM: C(M,N) = A(M,K)bf16 * B(K,N)f32 (+bias, epilogue)
// block tile 128x128, 8 waves, wave = 16x128 strip (8 wmma per K32 step).
// A tiles DMA'd by the Tensor Data Mover (async, double-buffered, overlapped
// with compute); B tiles converted f32->bf16 and stored TRANSPOSED so all
// fragments load as ds_load_b128. One barrier per K-step.
// modes: 2=QKV scatter  3=proj+unwindow+residual  4=GELU bf16  5=+xres f32
// ---------------------------------------------------------------------------
__global__ __launch_bounds__(256) void gemm_bf16_wmma(
    const unsigned short* __restrict__ A, const float* __restrict__ Bw,
    const float* __restrict__ bias, int M, int N, int K, int mode,
    unsigned short* __restrict__ out16, float* __restrict__ out32,
    const float* __restrict__ aux) {
  __shared__ unsigned short lA[2][128 * 32];    // 2 x 8 KB, row-major [m][k]
  __shared__ unsigned short lBt[2][128 * 32];   // 2 x 8 KB, transposed [n][k]

  int tid = threadIdx.x, lane = tid & 31, wid = tid >> 5;
  int bm = blockIdx.x * 128, bn = blockIdx.y * 128;
  int nn = lane & 15, mo = (lane >= 16) ? 8 : 0;

  v8f acc[8];
#pragma unroll
  for (int i = 0; i < 8; ++i) acc[i] = zero_v8f();

  int bkp = tid >> 4, bseg = (tid & 15) * 8;    // B-tile fill mapping
  int nk = K / 32;

  // prologue: TDM kicks off A tile 0
  if (wid == 0)
    tdm_load_tile_b16(A + (size_t)bm * K, (unsigned)(size_t)&lA[0][0],
                      32u, 128u, (unsigned)K);

  for (int i = 0; i < nk; ++i) {
    int k0 = i * 32;
    int cur = i & 1;
    {   // B tile 32x128 f32 -> bf16, transposed, packed b32 stores
      const float* s0 = Bw + (size_t)(k0 + 2 * bkp) * N + bn + bseg;
      const float* s1 = s0 + N;
#pragma unroll
      for (int t2 = 0; t2 < 8; ++t2)
        *(unsigned int*)(&lBt[cur][(bseg + t2) * 32 + 2 * bkp]) =
            pack2bf(s0[t2], s1[t2]);
    }
    if (i + 1 < nk)   // prefetch next B tile rows (global_prefetch_b8)
      __builtin_prefetch(Bw + (size_t)(k0 + 32 + 2 * bkp) * N + bn + bseg, 0, 1);

    if (wid == 0) __builtin_amdgcn_s_wait_tensorcnt(0);  // A[cur] landed
    __syncthreads();
    // all waves finished reading buffer (i+1)&1 in iteration i-1 -> refill it
    if (wid == 0 && i + 1 < nk)
      tdm_load_tile_b16(A + (size_t)bm * K + (k0 + 32),
                        (unsigned)(size_t)&lA[(i + 1) & 1][0],
                        32u, 128u, (unsigned)K);

    v16bf af;
#pragma unroll
    for (int j = 0; j < 8; ++j) {
      int kk = frag_k(j, lane);
      af[2 * j]     = us2bf(lA[cur][(wid * 16 + nn) * 32 + kk]);
      af[2 * j + 1] = us2bf(lA[cur][(wid * 16 + nn) * 32 + kk + 1]);
    }
#pragma unroll
    for (int nt = 0; nt < 8; ++nt) {
      v16bf bfr;
#pragma unroll
      for (int j = 0; j < 8; ++j) {
        int kk = frag_k(j, lane);
        bfr[2 * j]     = us2bf(lBt[cur][(nt * 16 + nn) * 32 + kk]);
        bfr[2 * j + 1] = us2bf(lBt[cur][(nt * 16 + nn) * 32 + kk + 1]);
      }
      acc[nt] = __builtin_amdgcn_wmma_f32_16x16x32_bf16(
          false, af, false, bfr, (short)0, acc[nt], false, false);
    }
  }

  // epilogue
#pragma unroll
  for (int nt = 0; nt < 8; ++nt) {
#pragma unroll
    for (int r = 0; r < 8; ++r) {
      int gm = bm + wid * 16 + r + mo;
      int gn = bn + nt * 16 + nn;
      float v = acc[nt][r];
      if (mode == 2) {            // QKV -> (win,{q,k,v},head,t,d) bf16
        v += bias[gn];
        int w_ = gm / LP, t = gm % LP;
        int s3 = gn / DIM, rem = gn % DIM, hd = rem >> 6, d = rem & 63;
        if (s3 == 1) v *= 0.125f; // fold HD^-0.5 into K
        out16[(((size_t)(w_ * 3 + s3) * NH + hd) * LP + t) * HD + d] = f2bf(v);
      } else if (mode == 3) {     // proj -> unwindow + residual (f32)
        int w_ = gm / LP, t = gm % LP;
        if (t < LQ) {
          int b_ = w_ / 25, wh = (w_ % 25) / 5, ww = w_ % 5;
          int hh = wh * WS14 + t / WS14, cc = ww * WS14 + t % WS14;
          if (hh < 64 && cc < 64) {
            size_t idx = (((size_t)b_ * 64 + hh) * 64 + cc) * DIM + gn;
            out32[idx] = v + bias[gn] + aux[idx];
          }
        }
      } else if (mode == 4) {     // MLP1: bias + exact GELU -> bf16
        v += bias[gn];
        v = 0.5f * v * (1.f + erff(v * 0.70710678118654752f));
        out16[(size_t)gm * N + gn] = f2bf(v);
      } else if (mode == 5) {     // MLP2: bias + residual -> f32 out
        out32[(size_t)gm * N + gn] = v + bias[gn] + aux[(size_t)gm * DIM + gn];
      } else {                    // plain bf16
        out16[(size_t)gm * N + gn] = f2bf(v);
      }
    }
  }
}

// ---------------------------------------------------------------------------
// Attention, one block per (window, head). q unscaled, k pre-scaled (1/8).
// Q & K fragments: direct global (per-lane contiguous -> global_load_b128,
// K tile is 28KB and L2-resident). V staged TRANSPOSED in LDS so P*V
// B-fragments are contiguous. Rel-pos bias tables transposed bf16 so the
// 8 query rows of a column-tile load as one 16B vector.
// ---------------------------------------------------------------------------
__global__ __launch_bounds__(256) void attn_wmma_kernel(
    const unsigned short* __restrict__ qkv, const float* __restrict__ rel_h_t,
    const float* __restrict__ rel_w_t, unsigned short* __restrict__ attnout) {
  int blk = blockIdx.x;
  int w_ = blk / NH, hd = blk % NH;
  const unsigned short* qb = qkv + ((size_t)(w_ * 3 + 0) * NH + hd) * LP * HD;
  const unsigned short* kb = qkv + ((size_t)(w_ * 3 + 1) * NH + hd) * LP * HD;
  const unsigned short* vb = qkv + ((size_t)(w_ * 3 + 2) * NH + hd) * LP * HD;

  __shared__ unsigned short svT[HD * LP];        // 28 KB, V transposed [d][tok]
  __shared__ unsigned short srelhT[WS14 * LP];   // 6.125 KB, bf16 [kh][t]
  __shared__ unsigned short srelwT[WS14 * LP];   // 6.125 KB, bf16 [kw][t]
  __shared__ unsigned short pstage[8][16 * 32];  // 8 KB

  int tid = threadIdx.x, lane = tid & 31, wid = tid >> 5;
  int nn = lane & 15, mo = (lane >= 16) ? 8 : 0;

  // V -> LDS transposed; coalesced 16B global reads, packed b32 LDS stores
  for (int it = tid; it < (LP / 2) * 8; it += 256) {
    int tp = it >> 3, g = it & 7;
    int tok = tp * 2;
    const unsigned short* v0 = vb + (size_t)tok * HD + g * 8;
    uint4 a = *(const uint4*)v0;
    uint4 b = *(const uint4*)(v0 + HD);
    unsigned short* pa = (unsigned short*)&a;
    unsigned short* pb = (unsigned short*)&b;
#pragma unroll
    for (int i = 0; i < 8; ++i) {
      unsigned int pk = (unsigned int)pa[i] | ((unsigned int)pb[i] << 16);
      *(unsigned int*)(&svT[(g * 8 + i) * LP + tok]) = pk;
    }
  }

  // decomposed rel-pos bias, transposed bf16: srelhT[kh][t] = dot(q[t], rel_h[ht-kh+13])
  for (int idx = tid; idx < WS14 * LP; idx += 256) {
    int kh = idx / LP, t = idx % LP;
    float ah = 0.f, aw = 0.f;
    if (t < LQ) {
      int ht = t / WS14, wt = t % WS14;
      const float* rh = rel_h_t + (size_t)(ht - kh + 13) * HD;
      const float* rw = rel_w_t + (size_t)(wt - kh + 13) * HD;
#pragma unroll 8
      for (int d = 0; d < HD; ++d) {
        float qv = bf2f(qb[(size_t)t * HD + d]);
        ah += qv * rh[d];
        aw += qv * rw[d];
      }
    }
    srelhT[idx] = f2bf(ah);
    srelwT[idx] = f2bf(aw);
  }
  __syncthreads();

  for (int tile = wid; tile < LP / 16; tile += 8) {
    // Q A-fragments (K = 0..31 and 32..63), per-lane contiguous global
    v16bf aq0, aq1;
    const unsigned short* qrow = qb + (size_t)(tile * 16 + nn) * HD;
#pragma unroll
    for (int j = 0; j < 8; ++j) {
      int kk = frag_k(j, lane);
      aq0[2 * j]     = us2bf(qrow[kk]);
      aq0[2 * j + 1] = us2bf(qrow[kk + 1]);
      aq1[2 * j]     = us2bf(qrow[kk + 32]);
      aq1[2 * j + 1] = us2bf(qrow[kk + 33]);
    }

    v8f sc[LP / 16];
#pragma unroll
    for (int nt = 0; nt < LP / 16; ++nt) {
      v8f a = zero_v8f();
      v16bf bk0, bk1;
      const unsigned short* krow = kb + (size_t)(nt * 16 + nn) * HD;
#pragma unroll
      for (int j = 0; j < 8; ++j) {
        int kk = frag_k(j, lane);
        bk0[2 * j]     = us2bf(krow[kk]);       // B[kk][n] = k[s][kk]
        bk0[2 * j + 1] = us2bf(krow[kk + 1]);
        bk1[2 * j]     = us2bf(krow[kk + 32]);
        bk1[2 * j + 1] = us2bf(krow[kk + 33]);
      }
      a = __builtin_amdgcn_wmma_f32_16x16x32_bf16(false, aq0, false, bk0,
                                                  (short)0, a, false, false);
      a = __builtin_amdgcn_wmma_f32_16x16x32_bf16(false, aq1, false, bk1,
                                                  (short)0, a, false, false);
      sc[nt] = a;
    }

    // bias (vectorized 16B LDS loads; rows t>=196 hold 0) + key mask
#pragma unroll
    for (int nt = 0; nt < LP / 16; ++nt) {
      int s = nt * 16 + nn;
      if (s < LQ) {
        int kh = s / WS14, kw = s % WS14;
        uint4 uh = *(const uint4*)(&srelhT[kh * LP + tile * 16 + mo]);
        uint4 uw = *(const uint4*)(&srelwT[kw * LP + tile * 16 + mo]);
        const unsigned short* ph = (const unsigned short*)&uh;
        const unsigned short* pw = (const unsigned short*)&uw;
#pragma unroll
        for (int r = 0; r < 8; ++r)
          sc[nt][r] += bf2f(ph[r]) + bf2f(pw[r]);
      } else {
#pragma unroll
        for (int r = 0; r < 8; ++r) sc[nt][r] = -1e30f;
      }
    }

    // softmax: row reductions within each 16-lane half (wave32)
    float rnorm[8];
#pragma unroll
    for (int r = 0; r < 8; ++r) {
      float mx = -1e30f;
#pragma unroll
      for (int nt = 0; nt < LP / 16; ++nt) mx = fmaxf(mx, sc[nt][r]);
      mx = fmaxf(mx, __shfl_xor(mx, 1, 32));
      mx = fmaxf(mx, __shfl_xor(mx, 2, 32));
      mx = fmaxf(mx, __shfl_xor(mx, 4, 32));
      mx = fmaxf(mx, __shfl_xor(mx, 8, 32));
      float sum = 0.f;
#pragma unroll
      for (int nt = 0; nt < LP / 16; ++nt) {
        float e = __expf(sc[nt][r] - mx);
        sc[nt][r] = e;
        sum += e;
      }
      sum += __shfl_xor(sum, 1, 32);
      sum += __shfl_xor(sum, 2, 32);
      sum += __shfl_xor(sum, 4, 32);
      sum += __shfl_xor(sum, 8, 32);
      rnorm[r] = 1.f / sum;
    }

    // O = P * V, P staged through per-wave LDS (C-layout -> A-layout)
    v8f oacc[4];
#pragma unroll
    for (int ct = 0; ct < 4; ++ct) oacc[ct] = zero_v8f();

#pragma unroll
    for (int kb2 = 0; kb2 < LP / 32; ++kb2) {
#pragma unroll
      for (int half = 0; half < 2; ++half) {
        int nt = kb2 * 2 + half;
#pragma unroll
        for (int r = 0; r < 8; ++r)
          pstage[wid][(r + mo) * 32 + half * 16 + nn] =
              f2bf(sc[nt][r] * rnorm[r]);
      }
      asm volatile("s_wait_dscnt 0x0" ::: "memory");   // cross-lane LDS RAW
      v16bf ap;
#pragma unroll
      for (int j = 0; j < 8; ++j) {
        int kk = frag_k(j, lane);
        ap[2 * j]     = us2bf(pstage[wid][nn * 32 + kk]);
        ap[2 * j + 1] = us2bf(pstage[wid][nn * 32 + kk + 1]);
      }
#pragma unroll
      for (int ct = 0; ct < 4; ++ct) {
        v16bf bv;
        const unsigned short* vcol = &svT[(ct * 16 + nn) * LP + kb2 * 32];
#pragma unroll
        for (int j = 0; j < 8; ++j) {
          int kk = frag_k(j, lane);
          bv[2 * j]     = us2bf(vcol[kk]);     // B[kk][n] = vT[d][tok]
          bv[2 * j + 1] = us2bf(vcol[kk + 1]);
        }
        oacc[ct] = __builtin_amdgcn_wmma_f32_16x16x32_bf16(
            false, ap, false, bv, (short)0, oacc[ct], false, false);
      }
    }

    // store O into windowed (MQ, 768) bf16 layout
#pragma unroll
    for (int ct = 0; ct < 4; ++ct) {
#pragma unroll
      for (int r = 0; r < 8; ++r) {
        int tq = tile * 16 + r + mo;
        int d = ct * 16 + nn;
        attnout[((size_t)w_ * LP + tq) * DIM + hd * HD + d] = f2bf(oacc[ct][r]);
      }
    }
  }
}

// ---------------------------------------------------------------------------
extern "C" void kernel_launch(void* const* d_in, const int* in_sizes, int n_in,
                              void* d_out, int out_size, void* d_ws,
                              size_t ws_size, hipStream_t stream) {
  (void)in_sizes; (void)n_in; (void)out_size; (void)ws_size;
  const float* x     = (const float*)d_in[0];
  const float* ln1w  = (const float*)d_in[1];
  const float* ln1b  = (const float*)d_in[2];
  const float* qkvw  = (const float*)d_in[3];
  const float* qkvb_ = (const float*)d_in[4];
  const float* projw = (const float*)d_in[5];
  const float* projb = (const float*)d_in[6];
  const float* relh  = (const float*)d_in[7];
  const float* relw  = (const float*)d_in[8];
  const float* ln2w  = (const float*)d_in[9];
  const float* ln2b  = (const float*)d_in[10];
  const float* w1    = (const float*)d_in[11];
  const float* b1    = (const float*)d_in[12];
  const float* w2    = (const float*)d_in[13];
  const float* b2    = (const float*)d_in[14];

  char* ws = (char*)d_ws;
  size_t off = 0;
  auto carve = [&](size_t bytes) {
    void* p = ws + off;
    off += (bytes + 255) & ~(size_t)255;
    return p;
  };
  unsigned short* hwin    = (unsigned short*)carve((size_t)MQ * DIM * 2);
  unsigned short* qkvbuf  = (unsigned short*)carve((size_t)NWIN * 3 * NH * LP * HD * 2);
  unsigned short* attnout = (unsigned short*)carve((size_t)MQ * DIM * 2);
  float*          xres    = (float*)carve((size_t)MT * DIM * 4);
  unsigned short* h2      = (unsigned short*)carve((size_t)MT * DIM * 2);
  unsigned short* mlph    = (unsigned short*)carve((size_t)MT * MLPH * 2);

  // 1) LN1 + window partition
  ln1_window_kernel<<<MQ, 256, 0, stream>>>(x, ln1w, ln1b, hwin);

  // 2) QKV GEMM (44800 x 2304 x 768), scatter + k-scale epilogue
  gemm_bf16_wmma<<<dim3(MQ / 128, 2304 / 128), 256, 0, stream>>>(
      hwin, qkvw, qkvb_, MQ, 3 * DIM, DIM, 2, qkvbuf, nullptr, nullptr);

  // 3) windowed attention with decomposed rel-pos bias
  attn_wmma_kernel<<<NWIN * NH, 256, 0, stream>>>(qkvbuf, relh, relw, attnout);

  // 4) proj GEMM + un-window + residual -> xres (f32)
  gemm_bf16_wmma<<<dim3(MQ / 128, DIM / 128), 256, 0, stream>>>(
      attnout, projw, projb, MQ, DIM, DIM, 3, nullptr, xres, x);

  // 5) LN2
  ln2_kernel<<<MT, 256, 0, stream>>>(xres, ln2w, ln2b, h2);

  // 6) MLP1 GEMM + exact GELU
  gemm_bf16_wmma<<<dim3(MT / 128, MLPH / 128), 256, 0, stream>>>(
      h2, w1, b1, MT, MLPH, DIM, 4, mlph, nullptr, nullptr);

  // 7) MLP2 GEMM + residual -> d_out (f32)
  gemm_bf16_wmma<<<dim3(MT / 128, DIM / 128), 256, 0, stream>>>(
      mlph, w2, b2, MT, DIM, MLPH, 5, nullptr, (float*)d_out, xres);
}
